// TransformerBlock_43980465111552
// MI455X (gfx1250) — compile-verified, matrix operand order checked by reading, not run on hardware
//
#include <hip/hip_runtime.h>
#include <hip/hip_bf16.h>

// ---- problem constants ----
#define BB  64
#define SS  1024
#define DD  128
#define HH  4
#define HD  32
#define HID 512
#define EPSV 1e-5f

typedef __attribute__((ext_vector_type(16))) __bf16 v16bf;
typedef __attribute__((ext_vector_type(8)))  float  v8f;

// Load a 16x32 bf16 WMMA fragment (A or B operand) from fp32 memory.
// Element (idx, k) lives at P[idx*stride_idx + k] (k-contiguous).
// Layout per CDNA5 ISA 7.12.2 (16-bit A-matrix 16x32):
//   lane<16 : idx = idx0+lane,    K = {k0..k0+7, k0+16..k0+23}
//   lane>=16: idx = idx0+lane-16, K = {k0+8..k0+15, k0+24..k0+31}
__device__ __forceinline__ v16bf load_frag(const float* __restrict__ P,
                                           long idx0, long stride_idx, long k0, int lane) {
    long idx = idx0 + (lane & 15);
    long kb  = k0 + (long)((lane >> 4) << 3);
    const float* base = P + idx * stride_idx + kb;
    v16bf f;
#pragma unroll
    for (int i = 0; i < 8; ++i)  f[i]     = (__bf16)base[i];        // 32B contiguous
#pragma unroll
    for (int i = 0; i < 8; ++i)  f[8 + i] = (__bf16)base[16 + i];   // 32B contiguous
    return f;
}

__device__ __forceinline__ v8f wmma_bf16(v16bf a, v16bf b, v8f c) {
    return __builtin_amdgcn_wmma_f32_16x16x32_bf16(false, a, false, b, (short)0, c,
                                                   false, false);
}

// ---- block reductions ----
__device__ __forceinline__ float block_sum_128(float v) {
    __shared__ float sm[4];
#pragma unroll
    for (int off = 16; off > 0; off >>= 1) v += __shfl_xor(v, off, 32);
    if ((threadIdx.x & 31) == 0) sm[threadIdx.x >> 5] = v;
    __syncthreads();
    float r = sm[0] + sm[1] + sm[2] + sm[3];
    __syncthreads();
    return r;
}

// ---- kernel: LN1 + key/query padding masks (one 128-thread block per row) ----
__global__ void ln1_mask_kernel(const float* __restrict__ seq,
                                const float* __restrict__ g1, const float* __restrict__ be1,
                                float* __restrict__ qin,
                                float* __restrict__ key_mask, float* __restrict__ query_mask) {
    long row = blockIdx.x;
    int t = threadIdx.x;
    float v = seq[row * DD + t];
    float s  = block_sum_128(v);
    float s2 = block_sum_128(v * v);
    float sa = block_sum_128(fabsf(v));
    float mu  = s * (1.f / DD);
    float var = s2 * (1.f / DD) - mu * mu;
    float rs  = rsqrtf(var + EPSV);
    float y = (v - mu) * rs * g1[t] + be1[t];
    qin[row * DD + t] = y;
    float sqa = block_sum_128(fabsf(y));
    if (t == 0) {
        key_mask[row]   = (sa  > 0.f) ? 1.f : 0.f;
        query_mask[row] = (sqa > 0.f) ? 1.f : 0.f;
    }
}

// ---- kernel: small weight transpose Wt[n*K+k] = W[k*N+n] ----
__global__ void transpose_w_kernel(const float* __restrict__ W, float* __restrict__ Wt,
                                   int K, int N) {
    long i = (long)blockIdx.x * 256 + threadIdx.x;
    if (i >= (long)K * N) return;
    long k = i / N, n = i - k * N;
    Wt[n * (long)K + k] = W[i];
}

// ---- kernel: V transpose [B,S,H,HD] -> [B,H,HD,S], LDS-tiled, coalesced both ways ----
// grid = (S/32, B*H); block = (32, 8)
__global__ void transpose_v_kernel(const float* __restrict__ Vb, float* __restrict__ Vt) {
    __shared__ float tile[32][33];
    int s0 = blockIdx.x * 32;
    int bh = blockIdx.y;
    int b = bh >> 2, h = bh & (HH - 1);
    int tx = threadIdx.x, ty = threadIdx.y;
    const float* src = Vb + (long)b * SS * DD + (long)h * HD;  // [s][d], d contiguous
#pragma unroll
    for (int i = 0; i < 4; ++i)
        tile[ty + 8 * i][tx] = src[(long)(s0 + ty + 8 * i) * DD + tx];  // tx = d
    __syncthreads();
    float* dst = Vt + ((long)bh * HD) * SS;                    // [d][s], s contiguous
#pragma unroll
    for (int i = 0; i < 4; ++i)
        dst[(long)(ty + 8 * i) * SS + s0 + tx] = tile[tx][ty + 8 * i];  // tx = s
}

// ---- generic bf16 WMMA GEMM, k-contiguous transposed weights ----
// C[M,N] = op(A[M,K] @ Wt[N,K]^T + bias) (+residual)(*rowmask)
// block = (32,4): 4 waves; each wave owns a 16x32 output tile (A-frag reused twice)
template <int RELU, int RES, int MASK>
__global__ void gemm_bf16t_kernel(const float* __restrict__ A, const float* __restrict__ Wt,
                                  const float* __restrict__ bias,
                                  const float* __restrict__ residual,
                                  const float* __restrict__ rowmask,
                                  float* __restrict__ C, int M, int N, int K) {
    int lane = threadIdx.x;
    long m0 = ((long)blockIdx.x * 4 + threadIdx.y) * 16;
    long n0 = (long)blockIdx.y * 32;
    v8f acc0 = {}, acc1 = {};
    for (int k0 = 0; k0 < K; k0 += 32) {
        if (k0 + 32 < K) __builtin_prefetch(A + m0 * K + k0 + 32, 0, 1);
        v16bf a  = load_frag(A,  m0,      K, k0, lane);
        v16bf b0 = load_frag(Wt, n0,      K, k0, lane);
        v16bf b1 = load_frag(Wt, n0 + 16, K, k0, lane);
        acc0 = wmma_bf16(a, b0, acc0);
        acc1 = wmma_bf16(a, b1, acc1);
    }
    long mb = m0 + (long)((lane >> 4) << 3);
#pragma unroll
    for (int t = 0; t < 2; ++t) {
        const v8f acc = t ? acc1 : acc0;
        long n = n0 + t * 16 + (lane & 15);
        float bn = bias[n];
#pragma unroll
        for (int r = 0; r < 8; ++r) {
            long m = mb + r;
            float v = acc[r] + bn;
            if (RELU) v = fmaxf(v, 0.f);
            if (RES)  v += residual[m * N + n];
            if (MASK) v *= rowmask[m];
            C[m * N + n] = v;
        }
    }
}

// ---- FUSED kernel: scores + causal/key-masked softmax for one 16-query strip ----
// grid = (S/16, B*H), block = (32, 8). The full 16x1024 fp32 score strip lives in
// LDS (64KB << 320KB/WGP), so raw scores never round-trip through HBM.
// Phase 1: 8 waves share one Q-frag; wave w computes causal k-tiles kt = w, w+8, ...
//          (one v_wmma each) and scatters scaled scores into the LDS strip.
// Phase 2: (row, sub) = (tid>>4, tid&15); 16-lane shuffle reductions for row max
//          and exp-sum; exp values cached back into the strip; final pass writes
//          full 1024-wide w rows (0 above diagonal, *query_mask/sum) to d_out.
__global__ __launch_bounds__(256) void scores_softmax_kernel(
        const float* __restrict__ Qb, const float* __restrict__ Kb,
        const float* __restrict__ key_mask, const float* __restrict__ query_mask,
        float* __restrict__ attn) {
    __shared__ float strip[16][SS];            // 64 KB
    int qt = blockIdx.x, bh = blockIdx.y;
    int b = bh >> 2, h = bh & (HH - 1);
    int lane = threadIdx.x, wave = threadIdx.y;
    const float* Qp = Qb + (long)b * SS * DD + (long)h * HD;  // Q[q*D + d]
    const float* Kp = Kb + (long)b * SS * DD + (long)h * HD;  // K[key*D + d]
    const float scale = 0.17677669529663687f;                 // 1/sqrt(32)

    // ---- phase 1: WMMA score tiles -> LDS strip ----
    v16bf a = load_frag(Qp, (long)qt * 16, DD, 0, lane);
    int mb = (lane >> 4) << 3;
    int nlo = lane & 15;
    for (int kt = wave; kt <= qt; kt += 8) {
        v16bf kf = load_frag(Kp, (long)kt * 16, DD, 0, lane);
        v8f c = {};
        c = wmma_bf16(a, kf, c);
        int n = kt * 16 + nlo;
#pragma unroll
        for (int r = 0; r < 8; ++r) strip[mb + r][n] = c[r] * scale;
    }
    __syncthreads();

    // ---- phase 2: row softmax out of LDS ----
    int tid = wave * 32 + lane;
    int row = tid >> 4;                // 0..15
    int sub = tid & 15;                // 16 threads per row
    int q = qt * 16 + row;
    const float* km = key_mask + (long)b * SS;

    float mx = -INFINITY;
#pragma unroll 4
    for (int j = 0; j < SS / 16; ++j) {
        int k = sub + 16 * j;
        bool valid = (k <= q) && (km[k] != 0.f);
        float v = valid ? strip[row][k] : -INFINITY;
        mx = fmaxf(mx, v);
    }
#pragma unroll
    for (int off = 8; off > 0; off >>= 1) mx = fmaxf(mx, __shfl_xor(mx, off, 16));

    float se = 0.f;
#pragma unroll 4
    for (int j = 0; j < SS / 16; ++j) {
        int k = sub + 16 * j;
        bool valid = (k <= q) && (km[k] != 0.f);
        float sv = strip[row][k];
        float e = valid ? __expf(sv - mx) : 0.f;
        strip[row][k] = e;
        se += e;
    }
#pragma unroll
    for (int off = 8; off > 0; off >>= 1) se += __shfl_xor(se, off, 16);

    float qm = query_mask[(long)b * SS + q];
    float inv = (se > 0.f) ? (qm / se) : 0.f;   // nan_to_num: dead rows -> 0
    float* wout = attn + ((long)bh * SS + q) * SS;
#pragma unroll 4
    for (int j = 0; j < SS / 16; ++j) {
        int k = sub + 16 * j;
        wout[k] = strip[row][k] * inv;
    }
}

// ---- kernel: attn_out = w @ V via transposed Vt; causal-truncated K loop ----
// grid = (S/16, B*H), block = 32; w-frag reused across both head-dim tiles
__global__ void attn_av_kernel(const float* __restrict__ attn, const float* __restrict__ Vt,
                               float* __restrict__ AO) {
    int qt = blockIdx.x, bh = blockIdx.y;
    int lane = threadIdx.x;
    int b = bh >> 2, h = bh & (HH - 1);
    const float* wrow = attn + (long)bh * SS * SS;      // w[q*S + k], k contiguous
    const float* Vtp  = Vt + ((long)bh * HD) * SS;      // Vt[d*S + key], key contiguous
    int kiters = (qt * 16 + 16 + 31) >> 5;              // w==0 above diagonal: over-read safe
    v8f acc0 = {}, acc1 = {};
    for (int kk = 0; kk < kiters; ++kk) {
        if (kk + 1 < kiters)
            __builtin_prefetch(wrow + (long)qt * 16 * SS + (kk + 1) * 32, 0, 1);
        v16bf a  = load_frag(wrow, (long)qt * 16, SS, (long)kk * 32, lane);
        v16bf v0 = load_frag(Vtp, 0,  SS, (long)kk * 32, lane);
        v16bf v1 = load_frag(Vtp, 16, SS, (long)kk * 32, lane);
        acc0 = wmma_bf16(a, v0, acc0);
        acc1 = wmma_bf16(a, v1, acc1);
    }
    float* out = AO + (long)b * SS * DD + (long)h * HD;
    long mb = (long)qt * 16 + (long)((lane >> 4) << 3);
#pragma unroll
    for (int t = 0; t < 2; ++t) {
        const v8f acc = t ? acc1 : acc0;
        long n = t * 16 + (lane & 15);
#pragma unroll
        for (int r = 0; r < 8; ++r) out[(mb + r) * DD + n] = acc[r];
    }
}

// ---- kernel: x = attn_out + q_in; xn = LN2(x) ----
__global__ void add_ln2_kernel(const float* __restrict__ AO, const float* __restrict__ qin,
                               const float* __restrict__ g2, const float* __restrict__ be2,
                               float* __restrict__ xn) {
    long row = blockIdx.x;
    int t = threadIdx.x;
    float v = AO[row * DD + t] + qin[row * DD + t];
    float s  = block_sum_128(v);
    float s2 = block_sum_128(v * v);
    float mu  = s * (1.f / DD);
    float var = s2 * (1.f / DD) - mu * mu;
    float rs  = rsqrtf(var + EPSV);
    xn[row * DD + t] = (v - mu) * rs * g2[t] + be2[t];
}

extern "C" void kernel_launch(void* const* d_in, const int* in_sizes, int n_in,
                              void* d_out, int out_size, void* d_ws, size_t ws_size,
                              hipStream_t stream) {
    (void)in_sizes; (void)n_in; (void)out_size; (void)ws_size;
    const float* seq  = (const float*)d_in[0];
    const float* mask = (const float*)d_in[1];
    const float* Wq = (const float*)d_in[2];
    const float* bq = (const float*)d_in[3];
    const float* Wk = (const float*)d_in[4];
    const float* bk = (const float*)d_in[5];
    const float* Wv = (const float*)d_in[6];
    const float* bv = (const float*)d_in[7];
    const float* g1 = (const float*)d_in[8];
    const float* be1 = (const float*)d_in[9];
    const float* g2 = (const float*)d_in[10];
    const float* be2 = (const float*)d_in[11];
    const float* W1 = (const float*)d_in[12];
    const float* bm1 = (const float*)d_in[13];
    const float* W2 = (const float*)d_in[14];
    const float* bm2 = (const float*)d_in[15];

    const long BSD = (long)BB * SS * DD;
    float* y    = (float*)d_out;
    float* attn = y + BSD;                       // [B,H,S,S] lives directly in d_out

    float* ws = (float*)d_ws;
    float* qin  = ws;  ws += BSD;
    float* Qb   = ws;  ws += BSD;
    float* Kb   = ws;  ws += BSD;
    float* Vb   = ws;  ws += BSD;
    float* Vt   = ws;  ws += BSD;                // V transposed [B,H,HD,S]
    float* AO   = ws;  ws += BSD;
    float* xn   = ws;  ws += BSD;
    float* hbuf = ws;  ws += (long)BB * SS * HID;
    float* Wqt  = ws;  ws += (long)DD * DD;
    float* Wkt  = ws;  ws += (long)DD * DD;
    float* Wvt  = ws;  ws += (long)DD * DD;
    float* W1t  = ws;  ws += (long)DD * HID;     // [HID, DD]
    float* W2t  = ws;  ws += (long)HID * DD;     // [DD, HID]
    float* key_mask   = ws;  ws += (long)BB * SS;
    float* query_mask = ws;

    const int M = BB * SS;  // 65536
    dim3 gblk(32, 4);

    // 0. transpose weights to k-contiguous layouts (tiny, one-shot)
    transpose_w_kernel<<<(DD * DD + 255) / 256, 256, 0, stream>>>(Wq, Wqt, DD, DD);
    transpose_w_kernel<<<(DD * DD + 255) / 256, 256, 0, stream>>>(Wk, Wkt, DD, DD);
    transpose_w_kernel<<<(DD * DD + 255) / 256, 256, 0, stream>>>(Wv, Wvt, DD, DD);
    transpose_w_kernel<<<(DD * HID + 255) / 256, 256, 0, stream>>>(W1, W1t, DD, HID);
    transpose_w_kernel<<<(HID * DD + 255) / 256, 256, 0, stream>>>(W2, W2t, HID, DD);

    // 1. LN1 + padding masks
    ln1_mask_kernel<<<M, 128, 0, stream>>>(seq, g1, be1, qin, key_mask, query_mask);

    // 2. Q/K/V projections (Q from normalized input, K/V from raw seq — SASRec quirk)
    gemm_bf16t_kernel<0,0,0><<<dim3(M / 64, DD / 32), gblk, 0, stream>>>(qin, Wqt, bq, nullptr, nullptr, Qb, M, DD, DD);
    gemm_bf16t_kernel<0,0,0><<<dim3(M / 64, DD / 32), gblk, 0, stream>>>(seq, Wkt, bk, nullptr, nullptr, Kb, M, DD, DD);
    gemm_bf16t_kernel<0,0,0><<<dim3(M / 64, DD / 32), gblk, 0, stream>>>(seq, Wvt, bv, nullptr, nullptr, Vb, M, DD, DD);

    // 3. V -> Vt [B,H,HD,S] for k-contiguous fragment loads in w@V
    transpose_v_kernel<<<dim3(SS / 32, BB * HH), dim3(32, 8), 0, stream>>>(Vb, Vt);

    // 4. fused scores + softmax: LDS-resident strip, single pass over HBM for w
    scores_softmax_kernel<<<dim3(SS / 16, BB * HH), dim3(32, 8), 0, stream>>>(
        Qb, Kb, key_mask, query_mask, attn);

    // 5. attn_out = w @ V
    attn_av_kernel<<<dim3(SS / 16, BB * HH), 32, 0, stream>>>(attn, Vt, AO);

    // 6. residual + LN2
    add_ln2_kernel<<<M, 128, 0, stream>>>(AO, qin, g2, be2, xn);

    // 7. FFN: h = relu(xn @ W1 + bm1); y = (h @ W2 + bm2 + xn) * mask
    gemm_bf16t_kernel<1,0,0><<<dim3(M / 64, HID / 32), gblk, 0, stream>>>(xn, W1t, bm1, nullptr, nullptr, hbuf, M, HID, DD);
    gemm_bf16t_kernel<0,1,1><<<dim3(M / 64, DD / 32), gblk, 0, stream>>>(hbuf, W2t, bm2, xn, mask, y, M, DD, HID);
}
